// RNetModel_58488864637422
// MI455X (gfx1250) — compile-verified, matrix operand order checked by reading, not run on hardware
//
#include <hip/hip_runtime.h>
#include <hip/hip_bf16.h>
#include <hip/hip_fp16.h>
#include <cstdint>

typedef __attribute__((ext_vector_type(16))) _Float16 v16h;
typedef __attribute__((ext_vector_type(8)))  float    v8f;

// ---------------- model dims ----------------
#define HH 75
#define BB 32
#define EE 300
#define LPASS 200
#define LQUES 40
#define H2 150
#define H3 225
#define H4 300

// ---------------- WMMA helpers (16x16x32 f16 -> f32) ----------------
// A fragment: 16x32 tile, row-major, lane r=lane&15 holds row M=r,
// K swizzle per ISA: k = (i>>3)*16 + ((lane>>4)<<3) + (i&7).
// B fragment: weights stored (N,Kp) row-major => same swizzle with row = N.
__device__ __forceinline__ v16h frag_ab(const _Float16* p, int ld) {
  int lane = threadIdx.x & 31;
  int r = lane & 15;
  int kb = (lane >> 4) << 3;
  v16h f;
#pragma unroll
  for (int i = 0; i < 16; ++i) {
    int k = ((i >> 3) << 4) + kb + (i & 7);
    f[i] = p[(size_t)r * ld + k];
  }
  return f;
}

__device__ __forceinline__ float act_apply(float v, int act) {
  if (act == 1) v = 1.f / (1.f + expf(-v));
  return v;
}
__device__ __forceinline__ float sigf(float x) { return 1.f / (1.f + expf(-x)); }

// C/D layout: lane n=lane&15 is column, VGPR i -> row m = i + ((lane>>4)<<3)
__device__ __forceinline__ void acc_store_lds(float* o, int ldo, v8f c,
                                              const float* bias, int nbase,
                                              int Nreal, int act) {
  int lane = threadIdx.x & 31;
  int n = lane & 15;
  int mb = (lane >> 4) << 3;
  int gn = nbase + n;
  float bv = (bias && gn < Nreal) ? bias[gn] : 0.f;
#pragma unroll
  for (int i = 0; i < 8; ++i) {
    o[(size_t)(mb + i) * ldo + n] = act_apply(c[i] + bv, act);
  }
}

// Cooperative M=32 x N GEMM: O(32 x N) = act(A(32 x Kp) @ W(N x Kp)^T + bias)
// A in LDS (f16, lda=Kp, zero-padded), W in global (f16, padded), O in LDS f32.
__device__ void mm_lds(const _Float16* A, int lda, const _Float16* W, int ldw,
                       float* O, int ldo, const float* bias, int Nreal,
                       int Ntiles, int Kc, int act) {
  int wave = threadIdx.x >> 5;
  int nw = blockDim.x >> 5;
  int tiles = 2 * Ntiles;
  for (int t = wave; t < tiles; t += nw) {
    int mt = t & 1, nt = t >> 1;
    v8f acc = {};
    for (int kc = 0; kc < Kc; ++kc) {
      v16h a = frag_ab(A + (size_t)mt * 16 * lda + kc * 32, lda);
      v16h b = frag_ab(W + (size_t)nt * 16 * ldw + kc * 32, ldw);
      acc = __builtin_amdgcn_wmma_f32_16x16x32_f16(false, a, false, b,
                                                   (short)0, acc, false, false);
    }
    acc_store_lds(O + (size_t)mt * 16 * ldo + nt * 16, ldo, acc, bias, nt * 16,
                  Nreal, act);
  }
}

// ---------------- async global->LDS staging (CDNA5 ASYNCcnt path) ---------
// Copies nbytes (multiple of 16, both pointers 16B aligned) from global to
// LDS using GLOBAL_LOAD_ASYNC_TO_LDS_B128. Caller must later execute
// async_wait() + __syncthreads() before reading the LDS region.
__device__ __forceinline__ void async_copy_g2l(void* ldsDst, const void* gSrc,
                                               int nbytes) {
  int chunks = nbytes >> 4;
  for (int c = threadIdx.x; c < chunks; c += blockDim.x) {
    unsigned la = (unsigned)(uintptr_t)((char*)ldsDst + c * 16);
    unsigned long long ga = (unsigned long long)(uintptr_t)((const char*)gSrc + c * 16);
    asm volatile("global_load_async_to_lds_b128 %0, %1, off"
                 :: "v"(la), "v"(ga) : "memory");
  }
}
__device__ __forceinline__ void async_wait() {
  asm volatile("s_wait_asynccnt 0x0" ::: "memory");
}

// ---------------- weight prep ----------------
__global__ void zero16_kernel(_Float16* p, size_t n) {
  size_t i = (size_t)blockIdx.x * blockDim.x + threadIdx.x;
  if (i < n) p[i] = (_Float16)0.f;
}
__global__ void cvt_pad_kernel(const float* __restrict__ src, int N, int K,
                               _Float16* __restrict__ dst, int ldd, int colOff) {
  int i = blockIdx.x * blockDim.x + threadIdx.x;
  if (i >= N * K) return;
  int r = i / K, c = i % K;
  dst[(size_t)r * ldd + colOff + c] = (_Float16)src[i];
}
__global__ void add_vec_kernel(const float* a, const float* b, float* o, int n) {
  int i = blockIdx.x * blockDim.x + threadIdx.x;
  if (i < n) o[i] = a[i] + b[i];
}

// ---------------- big time-parallel GEMM: Y(M x N) = A(M x K) @ W^T + b ----
__global__ void gemm_bias(const float* __restrict__ A, int M, int Kreal, int Kp,
                          const _Float16* __restrict__ W,
                          const float* __restrict__ bias, int Nreal,
                          float* __restrict__ Y, int ldy) {
  extern __shared__ char smem[];
  _Float16* as = (_Float16*)smem;  // 32 x Kp
  int m0 = blockIdx.x * 32;
  if (threadIdx.x == 0) __builtin_prefetch(W, 0, 1);  // global_prefetch_b8
  for (int i = threadIdx.x; i < 32 * Kp; i += blockDim.x) {
    int r = i / Kp, c = i % Kp;
    int m = m0 + r;
    float v = (m < M && c < Kreal) ? A[(size_t)m * Kreal + c] : 0.f;
    as[i] = (_Float16)v;
  }
  __syncthreads();
  int Ntiles = (Nreal + 15) >> 4;
  int Kc = Kp >> 5;
  int wave = threadIdx.x >> 5, nw = blockDim.x >> 5;
  int lane = threadIdx.x & 31;
  for (int t = wave; t < 2 * Ntiles; t += nw) {
    int mt = t & 1, nt = t >> 1;
    v8f acc = {};
    for (int kc = 0; kc < Kc; ++kc) {
      v16h a = frag_ab(as + (size_t)mt * 16 * Kp + kc * 32, Kp);
      v16h b = frag_ab(W + (size_t)nt * 16 * Kp + kc * 32, Kp);
      acc = __builtin_amdgcn_wmma_f32_16x16x32_f16(false, a, false, b,
                                                   (short)0, acc, false, false);
    }
    int n = nt * 16 + (lane & 15);
    int mb = (lane >> 4) << 3;
    if (n < Nreal) {
      float bv = bias ? bias[n] : 0.f;
#pragma unroll
      for (int i = 0; i < 8; ++i) {
        int m = m0 + mt * 16 + mb + i;
        if (m < M) Y[(size_t)m * ldy + n] = acc[i] + bv;
      }
    }
  }
}

// ---------------- persistent GRU scan (encoder) ----------------
struct EncArgs {
  const float* gi[4];      // (L,B,225) precomputed x@wi^T+bi
  float* y[4];             // output base, already column-offset, row stride 150
  int L[4];
  const _Float16* wh[2];   // (240 x 96) f16 padded
  const float* bh[2];      // 225
};

#define ENC_GI_OFF 47104
#define ENC_SMEM (47104 + BB * H3 * 4)  // + 28800-byte async gi tile

__global__ void enc_scan(EncArgs a) {
  extern __shared__ char smem[];
  _Float16* h16 = (_Float16*)smem;            // 32 x 96
  float* h32 = (float*)(smem + 6144);         // 32 x 80
  float* gh = (float*)(smem + 16384);         // 32 x 240
  float* giL = (float*)(smem + ENC_GI_OFF);   // 32 x 225 (async-staged gi[t])
  int id = blockIdx.x;
  const float* gi = a.gi[id];
  float* y = a.y[id];
  int L = a.L[id];
  int rev = id & 1;
  const _Float16* wh = a.wh[id & 1];
  const float* bh = a.bh[id & 1];
  int tid = threadIdx.x, bd = blockDim.x;
  for (int i = tid; i < 32 * 96; i += bd) h16[i] = (_Float16)0.f;
  for (int i = tid; i < 32 * 80; i += bd) h32[i] = 0.f;
  __syncthreads();
  for (int s = 0; s < L; ++s) {
    int t = rev ? (L - 1 - s) : s;
    // kick off async DMA of gi[t] into LDS; it overlaps the recurrent WMMA
    async_copy_g2l(giL, gi + (size_t)t * BB * H3, BB * H3 * 4);
    mm_lds(h16, 96, wh, 96, gh, 240, bh, H3, 15, 3, 0);
    async_wait();
    __syncthreads();
    for (int i = tid; i < BB * HH; i += bd) {
      int b = i / HH, j = i % HH;
      const float* gb = giL + b * H3;
      float r = sigf(gb[j] + gh[b * 240 + j]);
      float z = sigf(gb[HH + j] + gh[b * 240 + HH + j]);
      float n = tanhf(gb[2 * HH + j] + r * gh[b * 240 + 2 * HH + j]);
      float h = (1.f - z) * n + z * h32[b * 80 + j];
      h32[b * 80 + j] = h;
      h16[b * 96 + j] = (_Float16)h;
      y[(size_t)t * BB * H2 + b * H2 + j] = h;
    }
    __syncthreads();
  }
}

// ---------------- attention + gated GRU scan (coattention & self-match) ----
struct AttArgs {
  const float* pin;   // (L,B,150) input sequence
  const float* proj;  // (attLen,B,75) precomputed projection
  const float* mem;   // (attLen,B,150) context memory
  const float* vt;    // (B,75)
  const float* v0[2];
  const float* b_uv;      // 75 (combined bias)
  const _Float16* Wuv;    // 80 x 256  [Wp_u | Wp_v] (or [WPhat | 0])
  const float* bg;        // 300
  const _Float16* Wg;     // 304 x 320
  const float* bi[2];     // 225
  const _Float16* wi[2];  // 240 x 320
  const float* bh[2];     // 225
  const _Float16* wh[2];  // 240 x 96
  float* out;             // (L,B,150)
  int L, attLen, revProj, revMem;
};

__global__ void att_scan(AttArgs a) {
  extern __shared__ char smem[];
  _Float16* v16 = (_Float16*)smem;              // 32x96
  float* v32 = (float*)(smem + 6144);           // 32x80
  _Float16* astage = (_Float16*)(smem + 16384); // 32x320 f16
  float* u32 = (float*)(smem + 36864);          // 32x80
  float* s32 = (float*)(smem + 47104);          // 32x208
  float* cc32 = (float*)(smem + 73728);         // 32x304 (later: gi)
  float* gw = (float*)(smem + 112640);          // 32x304 (g, later gh)
  int br = blockIdx.x;
  const float* bi = a.bi[br];  const _Float16* wi = a.wi[br];
  const float* bh = a.bh[br];  const _Float16* wh = a.wh[br];
  float* out = a.out + br * HH;
  int tid = threadIdx.x, bd = blockDim.x;
  int AL = a.attLen;
  // init v
  for (int i = tid; i < 32 * 96; i += bd) {
    int b = i / 96, j = i % 96;
    float v = (j < HH) ? a.v0[br][b * HH + j] : 0.f;
    v16[i] = (_Float16)v;
    if (j < 80) v32[b * 80 + j] = (j < HH) ? v : 0.f;
  }
  __syncthreads();
  for (int s = 0; s < a.L; ++s) {
    const float* pt = a.pin + (size_t)(br ? (a.L - 1 - s) : s) * BB * H2;
    // u = [p_t | v] @ Wuv^T + b_uv
    for (int i = tid; i < 32 * 256; i += bd) {
      int b = i / 256, c = i % 256;
      float v = (c < H2) ? pt[b * H2 + c]
                         : ((c < H2 + HH) ? v32[b * 80 + (c - H2)] : 0.f);
      astage[i] = (_Float16)v;
    }
    __syncthreads();
    mm_lds(astage, 256, a.Wuv, 256, u32, 80, a.b_uv, HH, 5, 8, 0);
    __syncthreads();
    // scores
    for (int p = tid; p < BB * AL; p += bd) {
      int b = p & 31, l = p >> 5;
      int lp = (br && a.revProj) ? (AL - 1 - l) : l;
      const float* pr = a.proj + (size_t)lp * BB * HH + b * HH;
      const float* vtb = a.vt + b * HH;
      float sc = 0.f;
      for (int j = 0; j < HH; ++j) sc += tanhf(pr[j] + u32[b * 80 + j]) * vtb[j];
      s32[b * 208 + l] = sc;
    }
    __syncthreads();
    if (tid < 32) {
      int b = tid; float mx = -1e30f;
      for (int l = 0; l < AL; ++l) mx = fmaxf(mx, s32[b * 208 + l]);
      float sum = 0.f;
      for (int l = 0; l < AL; ++l) { float e = expf(s32[b * 208 + l] - mx); s32[b * 208 + l] = e; sum += e; }
      float inv = 1.f / sum;
      for (int l = 0; l < AL; ++l) s32[b * 208 + l] *= inv;
    }
    __syncthreads();
    // context + cc = [p_t | c]
    for (int i = tid; i < BB * H2; i += bd) {
      int b = i / H2, j = i % H2;
      float acc = 0.f;
      for (int l = 0; l < AL; ++l) {
        int lm = (br && a.revMem) ? (AL - 1 - l) : l;
        acc += s32[b * 208 + l] * a.mem[(size_t)lm * BB * H2 + b * H2 + j];
      }
      cc32[b * 304 + H2 + j] = acc;
      cc32[b * 304 + j] = pt[b * H2 + j];
    }
    __syncthreads();
    // g = sigmoid(cc @ Wg^T + bg)
    for (int i = tid; i < 32 * 320; i += bd) {
      int b = i / 320, c = i % 320;
      astage[i] = (_Float16)((c < H4) ? cc32[b * 304 + c] : 0.f);
    }
    __syncthreads();
    mm_lds(astage, 320, a.Wg, 320, gw, 304, a.bg, H4, 19, 10, 1);
    __syncthreads();
    // cc2 = g * cc
    for (int i = tid; i < 32 * 320; i += bd) {
      int b = i / 320, c = i % 320;
      float v = (c < H4) ? gw[b * 304 + c] * cc32[b * 304 + c] : 0.f;
      astage[i] = (_Float16)v;
    }
    __syncthreads();
    // gi = cc2 @ wi^T + bi (into cc32), gh = v @ wh^T + bh (into gw)
    mm_lds(astage, 320, wi, 320, cc32, 304, bi, H3, 15, 10, 0);
    mm_lds(v16, 96, wh, 96, gw, 304, bh, H3, 15, 3, 0);
    __syncthreads();
    for (int i = tid; i < BB * HH; i += bd) {
      int b = i / HH, j = i % HH;
      float r = sigf(cc32[b * 304 + j] + gw[b * 304 + j]);
      float z = sigf(cc32[b * 304 + HH + j] + gw[b * 304 + HH + j]);
      float n = tanhf(cc32[b * 304 + 2 * HH + j] + r * gw[b * 304 + 2 * HH + j]);
      float h = (1.f - z) * n + z * v32[b * 80 + j];
      v32[b * 80 + j] = h;
      v16[b * 96 + j] = (_Float16)h;
      out[(size_t)s * BB * H2 + b * H2 + j] = h;
    }
    __syncthreads();
  }
}

// ---------------- pointer network ----------------
struct PtrArgs {
  const float* qproj2;   // (40,B,75)
  const float* quesEnc;  // (40,B,150)
  const float* hP;       // (200,B,150)
  const float* projP;    // (200,B,75)
  const float* VQr;      // 75
  const float* WQv_w;    // 75x75
  const float* WQv_b;    // 75
  const float* vt1;      // (B,75)
  const float* vt2;      // (B,75)
  const _Float16* Wah;   // 80x160
  const float* bah;      // 75
  const _Float16* wi;    // 464x160 (ans_rnn wi)
  const float* bi;       // 450
  const _Float16* wh;    // 464x160
  const float* bh;       // 450
  float* out;            // 2*32*200
};

__device__ void point_scores(const PtrArgs& a, const float* u32, float* s32) {
  for (int p = threadIdx.x; p < BB * LPASS; p += blockDim.x) {
    int b = p & 31, l = p >> 5;
    const float* pr = a.projP + (size_t)l * BB * HH + b * HH;
    float sc = 0.f;
    for (int j = 0; j < HH; ++j)
      sc += tanhf(pr[j] + u32[b * 80 + j]) * a.vt2[b * HH + j];
    s32[b * 208 + l] = sc;
  }
}
__device__ void softmax_rows(float* s32, int n) {
  if (threadIdx.x < 32) {
    int b = threadIdx.x; float mx = -1e30f;
    for (int l = 0; l < n; ++l) mx = fmaxf(mx, s32[b * 208 + l]);
    float sum = 0.f;
    for (int l = 0; l < n; ++l) { float e = expf(s32[b * 208 + l] - mx); s32[b * 208 + l] = e; sum += e; }
    float inv = 1.f / sum;
    for (int l = 0; l < n; ++l) s32[b * 208 + l] *= inv;
  }
}

__global__ void ptr_kernel(PtrArgs a) {
  extern __shared__ char smem[];
  float* uvq = (float*)smem;                    // 80
  float* s32 = (float*)(smem + 320);            // 32x208
  float* rq32 = (float*)(smem + 26944);         // 32x160
  _Float16* rq16 = (_Float16*)(smem + 47424);   // 32x160
  float* u32 = (float*)(smem + 57664);          // 32x80
  float* c32 = (float*)(smem + 67904);          // 32x160
  _Float16* c16 = (_Float16*)(smem + 88384);    // 32x160
  float* gi32 = (float*)(smem + 98624);         // 32x464
  float* gh32 = (float*)(smem + 158016);        // 32x464
  int tid = threadIdx.x, bd = blockDim.x;
  for (int j = tid; j < HH; j += bd) {
    float s = a.WQv_b[j];
    for (int k = 0; k < HH; ++k) s += a.WQv_w[j * HH + k] * a.VQr[k];
    uvq[j] = s;
  }
  __syncthreads();
  for (int p = tid; p < BB * LQUES; p += bd) {
    int b = p & 31, l = p >> 5;
    const float* pr = a.qproj2 + (size_t)l * BB * HH + b * HH;
    float sc = 0.f;
    for (int j = 0; j < HH; ++j) sc += tanhf(pr[j] + uvq[j]) * a.vt1[b * HH + j];
    s32[b * 208 + l] = sc;
  }
  __syncthreads();
  softmax_rows(s32, LQUES);
  __syncthreads();
  for (int i = tid; i < 32 * 160; i += bd) {
    int b = i / 160, j = i % 160;
    float v = 0.f;
    if (j < H2)
      for (int l = 0; l < LQUES; ++l)
        v += s32[b * 208 + l] * a.quesEnc[(size_t)l * BB * H2 + b * H2 + j];
    rq32[i] = v; rq16[i] = (_Float16)v;
  }
  __syncthreads();
  mm_lds(rq16, 160, a.Wah, 160, u32, 80, a.bah, HH, 5, 5, 0);
  __syncthreads();
  point_scores(a, u32, s32);
  __syncthreads();
  softmax_rows(s32, LPASS);
  __syncthreads();
  for (int p = tid; p < BB * LPASS; p += bd) {
    int b = p & 31, l = p >> 5;
    a.out[b * LPASS + l] = s32[b * 208 + l];
  }
  for (int i = tid; i < 32 * 160; i += bd) {
    int b = i / 160, j = i % 160;
    float v = 0.f;
    if (j < H2)
      for (int l = 0; l < LPASS; ++l)
        v += s32[b * 208 + l] * a.hP[(size_t)l * BB * H2 + b * H2 + j];
    c32[i] = v; c16[i] = (_Float16)v;
  }
  __syncthreads();
  // rQ2 = gru_cell(x=rQ, h=c)
  mm_lds(rq16, 160, a.wi, 160, gi32, 464, a.bi, 3 * H2, 29, 5, 0);
  mm_lds(c16, 160, a.wh, 160, gh32, 464, a.bh, 3 * H2, 29, 5, 0);
  __syncthreads();
  for (int i = tid; i < BB * H2; i += bd) {
    int b = i / H2, j = i % H2;
    float r = sigf(gi32[b * 464 + j] + gh32[b * 464 + j]);
    float z = sigf(gi32[b * 464 + H2 + j] + gh32[b * 464 + H2 + j]);
    float n = tanhf(gi32[b * 464 + 2 * H2 + j] + r * gh32[b * 464 + 2 * H2 + j]);
    rq32[b * 160 + j] = (1.f - z) * n + z * c32[b * 160 + j];
  }
  __syncthreads();
  for (int i = tid; i < 32 * 160; i += bd) {
    int j = i % 160;
    rq16[i] = (_Float16)((j < H2) ? rq32[i] : 0.f);
  }
  __syncthreads();
  mm_lds(rq16, 160, a.Wah, 160, u32, 80, a.bah, HH, 5, 5, 0);
  __syncthreads();
  point_scores(a, u32, s32);
  __syncthreads();
  softmax_rows(s32, LPASS);
  __syncthreads();
  for (int p = tid; p < BB * LPASS; p += bd) {
    int b = p & 31, l = p >> 5;
    a.out[BB * LPASS + b * LPASS + l] = s32[b * 208 + l];
  }
}

// ---------------- host orchestration ----------------
// d_in layout: passEmb, quesEmb, then jax pytree leaves of params
// (dicts flattened by sorted keys).
enum {
  IN_PASS = 0, IN_QUES = 1,
  C_VT = 2, C_WPU_B = 3, C_WPU_W = 4, C_WPV_B = 5, C_WPV_W = 6,
  C_WQU_B = 7, C_WQU_W = 8, C_WG_B = 9, C_WG_W = 10,
  C_RF_BH = 11, C_RF_BI = 12, C_RF_WH = 13, C_RF_WI = 14,
  C_RR_BH = 15, C_RR_BI = 16, C_RR_WH = 17, C_RR_WI = 18,
  C_V0F = 19, C_V0R = 20,
  ENC0 = 21,  // + l*8 + d*4 + {0:bh,1:bi,2:wh,3:wi}
  P_VQR = 45, P_VT1 = 46, P_VT2 = 47, P_WPH_B = 48, P_WPH_W = 49,
  P_WQU_B = 50, P_WQU_W = 51, P_WQV_B = 52, P_WQV_W = 53,
  P_WAH_B = 54, P_WAH_W = 55, P_AR_BH = 56, P_AR_BI = 57,
  P_AR_WH = 58, P_AR_WI = 59,
  S_VT = 60, S_WPV_B = 61, S_WPV_W = 62, S_WPH_B = 63, S_WPH_W = 64,
  S_WG_B = 65, S_WG_W = 66, S_H0F = 67, S_H0R = 68,
  S_RF_BH = 69, S_RF_BI = 70, S_RF_WH = 71, S_RF_WI = 72,
  S_RR_BH = 73, S_RR_BI = 74, S_RR_WH = 75, S_RR_WI = 76,
  N_IN_EXPECT = 77
};

extern "C" void kernel_launch(void* const* d_in, const int* in_sizes, int n_in,
                              void* d_out, int out_size, void* d_ws,
                              size_t ws_size, hipStream_t stream) {
  if (n_in < N_IN_EXPECT) return;
  auto P = [&](int i) { return (const float*)d_in[i]; };
  char* ws = (char*)d_ws;
  size_t off = 0;
  auto a16 = [&](size_t n) {
    off = (off + 255) & ~(size_t)255;
    _Float16* p = (_Float16*)(ws + off);
    off += n * sizeof(_Float16);
    return p;
  };
  auto a32 = [&](size_t n) {
    off = (off + 255) & ~(size_t)255;
    float* p = (float*)(ws + off);
    off += n * sizeof(float);
    return p;
  };

  // ---- f16 weight pool ----
  size_t poolStart = (off + 255) & ~(size_t)255;
  _Float16 *enc_wi[3][2], *enc_wh[3][2];
  for (int l = 0; l < 3; ++l)
    for (int d = 0; d < 2; ++d) {
      enc_wi[l][d] = a16((size_t)240 * (l == 0 ? 320 : 160));
      enc_wh[l][d] = a16((size_t)240 * 96);
    }
  _Float16* co_WQu = a16(80 * 160);
  _Float16* co_Wuv = a16(80 * 256);
  _Float16* co_Wg = a16((size_t)304 * 320);
  _Float16 *co_wi[2], *co_wh[2];
  for (int d = 0; d < 2; ++d) { co_wi[d] = a16((size_t)240 * 320); co_wh[d] = a16((size_t)240 * 96); }
  _Float16* sm_WPv = a16(80 * 160);
  _Float16* sm_Wuv = a16(80 * 256);
  _Float16* sm_Wg = a16((size_t)304 * 320);
  _Float16 *sm_wi[2], *sm_wh[2];
  for (int d = 0; d < 2; ++d) { sm_wi[d] = a16((size_t)240 * 320); sm_wh[d] = a16((size_t)240 * 96); }
  _Float16* pt_WQu = a16(80 * 160);
  _Float16* pt_WPh = a16(80 * 160);
  _Float16* pt_Wah = a16(80 * 160);
  _Float16* pt_wi = a16((size_t)464 * 160);
  _Float16* pt_wh = a16((size_t)464 * 160);
  size_t poolEnd = off;

  // ---- f32 scratch ----
  float* b_uv = a32(HH);
  float* giP[2]; float* giQ[2];
  giP[0] = a32((size_t)LPASS * BB * H3); giP[1] = a32((size_t)LPASS * BB * H3);
  giQ[0] = a32((size_t)LQUES * BB * H3); giQ[1] = a32((size_t)LQUES * BB * H3);
  float* encP0 = a32((size_t)LPASS * BB * H2);
  float* encP1 = a32((size_t)LPASS * BB * H2);
  float* encQ0 = a32((size_t)LQUES * BB * H2);
  float* encQ1 = a32((size_t)LQUES * BB * H2);
  float* qproj = a32((size_t)LQUES * BB * HH);
  float* vP = a32((size_t)LPASS * BB * H2);
  float* projf = a32((size_t)LPASS * BB * HH);
  float* hP = a32((size_t)LPASS * BB * H2);
  float* qproj2 = a32((size_t)LQUES * BB * HH);
  float* projP = a32((size_t)LPASS * BB * HH);
  if (off > ws_size) return;

  // ---- prep: zero pool, convert/pad weights to f16 ----
  size_t halves = (poolEnd - poolStart) / 2;
  zero16_kernel<<<(unsigned)((halves + 255) / 256), 256, 0, stream>>>(
      (_Float16*)(ws + poolStart), halves);
  auto cvt = [&](const float* src, int N, int K, _Float16* dst, int ldd, int co) {
    int n = N * K;
    cvt_pad_kernel<<<(n + 255) / 256, 256, 0, stream>>>(src, N, K, dst, ldd, co);
  };
  for (int l = 0; l < 3; ++l)
    for (int d = 0; d < 2; ++d) {
      int base = ENC0 + l * 8 + d * 4;
      cvt(P(base + 3), H3, (l == 0 ? EE : H2), enc_wi[l][d], (l == 0 ? 320 : 160), 0);
      cvt(P(base + 2), H3, HH, enc_wh[l][d], 96, 0);
    }
  cvt(P(C_WQU_W), HH, H2, co_WQu, 160, 0);
  cvt(P(C_WPU_W), HH, H2, co_Wuv, 256, 0);
  cvt(P(C_WPV_W), HH, HH, co_Wuv, 256, H2);
  cvt(P(C_WG_W), H4, H4, co_Wg, 320, 0);
  cvt(P(C_RF_WI), H3, H4, co_wi[0], 320, 0);
  cvt(P(C_RR_WI), H3, H4, co_wi[1], 320, 0);
  cvt(P(C_RF_WH), H3, HH, co_wh[0], 96, 0);
  cvt(P(C_RR_WH), H3, HH, co_wh[1], 96, 0);
  cvt(P(S_WPV_W), HH, H2, sm_WPv, 160, 0);
  cvt(P(S_WPH_W), HH, H2, sm_Wuv, 256, 0);  // WPhat_v; v-columns stay 0
  cvt(P(S_WG_W), H4, H4, sm_Wg, 320, 0);
  cvt(P(S_RF_WI), H3, H4, sm_wi[0], 320, 0);
  cvt(P(S_RR_WI), H3, H4, sm_wi[1], 320, 0);
  cvt(P(S_RF_WH), H3, HH, sm_wh[0], 96, 0);
  cvt(P(S_RR_WH), H3, HH, sm_wh[1], 96, 0);
  cvt(P(P_WQU_W), HH, H2, pt_WQu, 160, 0);
  cvt(P(P_WPH_W), HH, H2, pt_WPh, 160, 0);
  cvt(P(P_WAH_W), HH, H2, pt_Wah, 160, 0);
  cvt(P(P_AR_WI), 3 * H2, H2, pt_wi, 160, 0);
  cvt(P(P_AR_WH), 3 * H2, H2, pt_wh, 160, 0);
  add_vec_kernel<<<1, 128, 0, stream>>>(P(C_WPU_B), P(C_WPV_B), b_uv, HH);

  // ---- encoder: 3 bidirectional GRU layers ----
  const float* inP = P(IN_PASS);
  const float* inQ = P(IN_QUES);
  float* outP[3] = {encP0, encP1, encP0};
  float* outQ[3] = {encQ0, encQ1, encQ0};
  for (int l = 0; l < 3; ++l) {
    int K = (l == 0) ? EE : H2;
    int Kp = (l == 0) ? 320 : 160;
    size_t sh = (size_t)32 * Kp * 2;
    for (int d = 0; d < 2; ++d) {
      int base = ENC0 + l * 8 + d * 4;
      gemm_bias<<<(LPASS * BB + 31) / 32, 256, sh, stream>>>(
          inP, LPASS * BB, K, Kp, enc_wi[l][d], P(base + 1), H3, giP[d], H3);
      gemm_bias<<<(LQUES * BB + 31) / 32, 256, sh, stream>>>(
          inQ, LQUES * BB, K, Kp, enc_wi[l][d], P(base + 1), H3, giQ[d], H3);
    }
    EncArgs ea;
    ea.gi[0] = giP[0]; ea.gi[1] = giP[1]; ea.gi[2] = giQ[0]; ea.gi[3] = giQ[1];
    ea.y[0] = outP[l]; ea.y[1] = outP[l] + HH;
    ea.y[2] = outQ[l]; ea.y[3] = outQ[l] + HH;
    ea.L[0] = LPASS; ea.L[1] = LPASS; ea.L[2] = LQUES; ea.L[3] = LQUES;
    int base = ENC0 + l * 8;
    ea.wh[0] = enc_wh[l][0]; ea.wh[1] = enc_wh[l][1];
    ea.bh[0] = P(base + 0); ea.bh[1] = P(base + 4 + 0);
    enc_scan<<<4, 256, ENC_SMEM, stream>>>(ea);
    inP = outP[l]; inQ = outQ[l];
  }
  const float* passEnc = inP;  // encP0
  const float* quesEnc = inQ;  // encQ0

  // ---- coattention ----
  gemm_bias<<<(LQUES * BB + 31) / 32, 256, 32 * 160 * 2, stream>>>(
      quesEnc, LQUES * BB, H2, 160, co_WQu, P(C_WQU_B), HH, qproj, HH);
  AttArgs ca;
  ca.pin = passEnc; ca.proj = qproj; ca.mem = quesEnc; ca.vt = P(C_VT);
  ca.v0[0] = P(C_V0F); ca.v0[1] = P(C_V0R);
  ca.b_uv = b_uv; ca.Wuv = co_Wuv; ca.bg = P(C_WG_B); ca.Wg = co_Wg;
  ca.bi[0] = P(C_RF_BI); ca.bi[1] = P(C_RR_BI);
  ca.wi[0] = co_wi[0]; ca.wi[1] = co_wi[1];
  ca.bh[0] = P(C_RF_BH); ca.bh[1] = P(C_RR_BH);
  ca.wh[0] = co_wh[0]; ca.wh[1] = co_wh[1];
  ca.out = vP; ca.L = LPASS; ca.attLen = LQUES; ca.revProj = 0; ca.revMem = 0;
  att_scan<<<2, 256, 151552, stream>>>(ca);

  // ---- self-match ----
  gemm_bias<<<(LPASS * BB + 31) / 32, 256, 32 * 160 * 2, stream>>>(
      vP, LPASS * BB, H2, 160, sm_WPv, P(S_WPV_B), HH, projf, HH);
  AttArgs sa;
  sa.pin = vP; sa.proj = projf; sa.mem = vP; sa.vt = P(S_VT);
  sa.v0[0] = P(S_H0F); sa.v0[1] = P(S_H0R);
  sa.b_uv = P(S_WPH_B); sa.Wuv = sm_Wuv; sa.bg = P(S_WG_B); sa.Wg = sm_Wg;
  sa.bi[0] = P(S_RF_BI); sa.bi[1] = P(S_RR_BI);
  sa.wi[0] = sm_wi[0]; sa.wi[1] = sm_wi[1];
  sa.bh[0] = P(S_RF_BH); sa.bh[1] = P(S_RR_BH);
  sa.wh[0] = sm_wh[0]; sa.wh[1] = sm_wh[1];
  sa.out = hP; sa.L = LPASS; sa.attLen = LPASS; sa.revProj = 1; sa.revMem = 1;
  att_scan<<<2, 256, 151552, stream>>>(sa);

  // ---- pointer ----
  gemm_bias<<<(LQUES * BB + 31) / 32, 256, 32 * 160 * 2, stream>>>(
      quesEnc, LQUES * BB, H2, 160, pt_WQu, P(P_WQU_B), HH, qproj2, HH);
  gemm_bias<<<(LPASS * BB + 31) / 32, 256, 32 * 160 * 2, stream>>>(
      hP, LPASS * BB, H2, 160, pt_WPh, P(P_WPH_B), HH, projP, HH);
  PtrArgs pa;
  pa.qproj2 = qproj2; pa.quesEnc = quesEnc; pa.hP = hP; pa.projP = projP;
  pa.VQr = P(P_VQR); pa.WQv_w = P(P_WQV_W); pa.WQv_b = P(P_WQV_B);
  pa.vt1 = P(P_VT1); pa.vt2 = P(P_VT2);
  pa.Wah = pt_Wah; pa.bah = P(P_WAH_B);
  pa.wi = pt_wi; pa.bi = P(P_AR_BI);
  pa.wh = pt_wh; pa.bh = P(P_AR_BH);
  pa.out = (float*)d_out;
  ptr_kernel<<<1, 256, 217408, stream>>>(pa);
  (void)in_sizes; (void)out_size;
}
